// LatticeGNN_24635932410338
// MI455X (gfx1250) — compile-verified, matrix operand order checked by reading, not run on hardware
//
#include <hip/hip_runtime.h>
#include <hip/hip_bf16.h>
#include <stdint.h>

// ---------------------------------------------------------------------------
// Types for CDNA5 WMMA (wave32, v_wmma_f32_16x16x32_bf16)
// ---------------------------------------------------------------------------
typedef __bf16 v16bf __attribute__((ext_vector_type(16)));
typedef __bf16 v8bf  __attribute__((ext_vector_type(8)));
typedef float  v8f   __attribute__((ext_vector_type(8)));
typedef int    v4i   __attribute__((vector_size(16)));   // b128 payload type

#define GAS __attribute__((address_space(1)))   // global
#define LAS __attribute__((address_space(3)))   // LDS

#define LDT (32 + 8)   // LDS tile row stride (bf16 elems); 80B, 16B-aligned chunks

// gfx1250 async global->LDS staging (ASYNCcnt path); guarded fallback to the
// sync path if the builtin is absent.
#if defined(__HIP_DEVICE_COMPILE__) && defined(__gfx1250__) && \
    __has_builtin(__builtin_amdgcn_global_load_async_to_lds_b128)
#define HAS_ASYNC_LDS 1
#else
#define HAS_ASYNC_LDS 0
#endif

__device__ __forceinline__ unsigned short f32_to_bf16(float f) {
  unsigned int u = __float_as_uint(f);
  u += 0x7FFFu + ((u >> 16) & 1u);     // round-to-nearest-even
  return (unsigned short)(u >> 16);
}

// ---------------------------------------------------------------------------
// f32 -> bf16 elementwise (weights + activations)
// ---------------------------------------------------------------------------
__global__ void k_f32_to_bf16(const float* __restrict__ src,
                              unsigned short* __restrict__ dst, long long n) {
  long long i = (long long)blockIdx.x * blockDim.x + threadIdx.x;
  if (i < n) dst[i] = f32_to_bf16(src[i]);
}

// ---------------------------------------------------------------------------
// per-destination degree (float, for mean)
// ---------------------------------------------------------------------------
__global__ void k_degree(const long long* __restrict__ dst,
                         float* __restrict__ deg, int E) {
  int e = blockIdx.x * blockDim.x + threadIdx.x;
  if (e < E) atomicAdd(&deg[(int)dst[e]], 1.0f);
}

// ---------------------------------------------------------------------------
// edge scatter-add: agg[dst] += h[src]   (one block per edge, 4 floats/thread)
// ---------------------------------------------------------------------------
__global__ void k_scatter_add(const float* __restrict__ h,
                              const long long* __restrict__ src,
                              const long long* __restrict__ dst,
                              float* __restrict__ agg, int D) {
  const int e = blockIdx.x;
  const int d = threadIdx.x * 4;
  const long long s = src[e], t = dst[e];
  const float4 v = *(const float4*)(h + (long long)s * D + d);
  float* a = agg + (long long)t * D + d;
  atomicAdd(a + 0, v.x);
  atomicAdd(a + 1, v.y);
  atomicAdd(a + 2, v.z);
  atomicAdd(a + 3, v.w);
}

// ---------------------------------------------------------------------------
// mean = agg/max(deg,1) -> bf16 ; also h -> bf16 ; zero-pads rows N..Npad
// ---------------------------------------------------------------------------
__global__ void k_mean_convert(const float* __restrict__ agg,
                               const float* __restrict__ deg,
                               const float* __restrict__ h,
                               unsigned short* __restrict__ mbf,
                               unsigned short* __restrict__ hbf,
                               int N, int D) {
  const long long n = blockIdx.x;
  const int d = threadIdx.x * 4;
  ushort4 mo, ho;
  if (n < N) {
    const float inv = 1.0f / fmaxf(deg[n], 1.0f);
    const float4 a = *(const float4*)(agg + n * D + d);
    const float4 x = *(const float4*)(h + n * D + d);
    mo = make_ushort4(f32_to_bf16(a.x * inv), f32_to_bf16(a.y * inv),
                      f32_to_bf16(a.z * inv), f32_to_bf16(a.w * inv));
    ho = make_ushort4(f32_to_bf16(x.x), f32_to_bf16(x.y),
                      f32_to_bf16(x.z), f32_to_bf16(x.w));
  } else {
    mo = make_ushort4(0, 0, 0, 0);
    ho = mo;
  }
  *(ushort4*)(mbf + n * D + d) = mo;
  *(ushort4*)(hbf + n * D + d) = ho;
}

// ---------------------------------------------------------------------------
// Fused dual GEMM:  out = leaky( A1 @ B1^T + A2 @ B2^T + bias )
//   A1/A2 : [Npad, K]  bf16 row-major (padded rows are zero)
//   B1/B2 : [Dout, K]  bf16 row-major (weights; B = W^T -> contiguous rows)
// Block: 256 threads = 8 wave32s in a 4(M)x2(N) grid, block tile 128x128,
// K-step 32, 8 WMMA accumulators (32x64) per wave.
// Double-buffered LDS: the async-to-LDS DMA for tile k+1 is issued right
// after the barrier that publishes tile k, and ASYNCcnt is only waited at the
// top of the next iteration -> HBM latency hides behind the 8 WMMAs, and
// only one barrier per K-step is needed.
// ---------------------------------------------------------------------------
__launch_bounds__(256)
__global__ void k_sage_gemm(const unsigned short* __restrict__ A1,
                            const unsigned short* __restrict__ B1,
                            const unsigned short* __restrict__ A2,
                            const unsigned short* __restrict__ B2,
                            const float* __restrict__ bias,
                            float* __restrict__ out,
                            int Nreal, int K, int ldo) {
  __shared__ __align__(16) unsigned short lA[2][128 * LDT];  // 2x 128x32 A tile
  __shared__ __align__(16) unsigned short lB[2][128 * LDT];  // 2x 128x32 B tile

  const int tid  = threadIdx.x;
  const int lane = tid & 31;
  const int wave = tid >> 5;
  const int ln   = lane & 15;        // position within half-wave
  const int hi   = lane >> 4;        // 0: K 0-7/16-23, 1: K 8-15/24-31
  const int wm   = wave >> 1;        // 0..3  (32 rows each)
  const int wn   = wave & 1;         // 0..1  (64 cols each)
  const long long blockM = (long long)blockIdx.y * 128;
  const int       blockN = blockIdx.x * 128;

  // tile staging: 128 rows x 32 elems; 2 threads/row, 16 elems (2x16B) each
  const int srow = tid >> 1, schunk = (tid & 1) * 16;

  const int KT    = K >> 5;       // K-steps per pass
  const int STEPS = 2 * KT;       // pass 0: A1/B1, pass 1: A2/B2

  const unsigned short* aBase0 = A1 + (blockM + srow) * K + schunk;
  const unsigned short* aBase1 = A2 + (blockM + srow) * K + schunk;
  const unsigned short* bBase0 = B1 + (long long)(blockN + srow) * K + schunk;
  const unsigned short* bBase1 = B2 + (long long)(blockN + srow) * K + schunk;

  // issue staging (async DMA or sync fallback) for pipeline step `step`
  auto stage = [&](int step) {
    const int pass = (step >= KT) ? 1 : 0;
    const int kt   = step - pass * KT;
    const unsigned short* aSrc = (pass ? aBase1 : aBase0) + kt * 32;
    const unsigned short* bSrc = (pass ? bBase1 : bBase0) + kt * 32;
    unsigned short* aDst = &lA[step & 1][srow * LDT + schunk];
    unsigned short* bDst = &lB[step & 1][srow * LDT + schunk];
#if HAS_ASYNC_LDS
    __builtin_amdgcn_global_load_async_to_lds_b128(
        (GAS v4i*)(aSrc), (LAS v4i*)(aDst), 0, 0);
    __builtin_amdgcn_global_load_async_to_lds_b128(
        (GAS v4i*)(aSrc + 8), (LAS v4i*)(aDst + 8), 0, 0);
    __builtin_amdgcn_global_load_async_to_lds_b128(
        (GAS v4i*)(bSrc), (LAS v4i*)(bDst), 0, 0);
    __builtin_amdgcn_global_load_async_to_lds_b128(
        (GAS v4i*)(bSrc + 8), (LAS v4i*)(bDst + 8), 0, 0);
#else
    const uint4 av0 = *(const uint4*)(aSrc);
    const uint4 av1 = *(const uint4*)(aSrc + 8);
    const uint4 bv0 = *(const uint4*)(bSrc);
    const uint4 bv1 = *(const uint4*)(bSrc + 8);
    *(uint4*)(aDst)     = av0;
    *(uint4*)(aDst + 8) = av1;
    *(uint4*)(bDst)     = bv0;
    *(uint4*)(bDst + 8) = bv1;
#endif
    if (step + 1 < STEPS) {                   // global_prefetch_b8 next tile
      __builtin_prefetch(aSrc + 32, 0, 1);
      __builtin_prefetch(bSrc + 32, 0, 1);
    }
  };

  v8f acc[2][4] = {};

  stage(0);                                    // prime the pipeline
  for (int step = 0; step < STEPS; ++step) {
#if HAS_ASYNC_LDS
#if __has_builtin(__builtin_amdgcn_s_wait_asynccnt)
    __builtin_amdgcn_s_wait_asynccnt(0);       // my DMA writes for this tile
#else
    asm volatile("s_wait_asynccnt 0" ::: "memory");
#endif
#endif
    __syncthreads();                           // everyone's tile `step` ready;
                                               // reads of buf step^1 all done
    if (step + 1 < STEPS) stage(step + 1);     // overlap DMA with WMMAs below

    const int buf = step & 1;
    // fragments per ISA 16-bit A 16x32 layout (two 16B chunks per lane)
    v16bf aF[2], bF[4];
#pragma unroll
    for (int t = 0; t < 2; ++t) {
      const unsigned short* pa = &lA[buf][(wm * 32 + t * 16 + ln) * LDT];
      *(v8bf*)&aF[t]       = *(const v8bf*)(pa + hi * 8);
      *((v8bf*)&aF[t] + 1) = *(const v8bf*)(pa + 16 + hi * 8);
    }
#pragma unroll
    for (int t = 0; t < 4; ++t) {
      const unsigned short* pb = &lB[buf][(wn * 64 + t * 16 + ln) * LDT];
      *(v8bf*)&bF[t]       = *(const v8bf*)(pb + hi * 8);
      *((v8bf*)&bF[t] + 1) = *(const v8bf*)(pb + 16 + hi * 8);
    }

#pragma unroll
    for (int tm = 0; tm < 2; ++tm)
#pragma unroll
      for (int tn = 0; tn < 4; ++tn)
        acc[tm][tn] = __builtin_amdgcn_wmma_f32_16x16x32_bf16(
            false, aF[tm], false, bF[tn], (short)0, acc[tm][tn],
            false, false);
  }

  // epilogue: bias + leaky_relu, f32 C/D layout (VGPR r -> M=r / M=r+8)
#pragma unroll
  for (int tm = 0; tm < 2; ++tm) {
#pragma unroll
    for (int tn = 0; tn < 4; ++tn) {
      const int nOut = blockN + wn * 64 + tn * 16 + ln;
      const float b = bias[nOut];
#pragma unroll
      for (int r = 0; r < 8; ++r) {
        const long long m = blockM + wm * 32 + tm * 16 + hi * 8 + r;
        if (m < Nreal) {
          float v = acc[tm][tn][r] + b;
          v = (v >= 0.0f) ? v : 0.01f * v;
          out[m * ldo + nOut] = v;
        }
      }
    }
  }
}

// ---------------------------------------------------------------------------
// final projection: out[n] = h[n,:] . Wout + bout  (one wave32 per node)
// ---------------------------------------------------------------------------
__global__ void k_outproj(const float* __restrict__ h,
                          const float* __restrict__ w,
                          const float* __restrict__ b,
                          float* __restrict__ out, int N, int D) {
  const int lane = threadIdx.x & 31;
  const long long node =
      (((long long)blockIdx.x * blockDim.x + threadIdx.x) >> 5);
  if (node >= N) return;
  const float* hp = h + node * D;
  float s = 0.0f;
  for (int d = lane; d < D; d += 32) s = fmaf(hp[d], w[d], s);
#pragma unroll
  for (int off = 16; off > 0; off >>= 1) s += __shfl_down(s, off);
  if (lane == 0) out[node] = s + b[0];
}

// ---------------------------------------------------------------------------
// host launch
// ---------------------------------------------------------------------------
extern "C" void kernel_launch(void* const* d_in, const int* in_sizes, int n_in,
                              void* d_out, int out_size, void* d_ws,
                              size_t ws_size, hipStream_t stream) {
  const float*     x    = (const float*)d_in[0];
  const long long* ei   = (const long long*)d_in[1];
  const float*     Wl   = (const float*)d_in[2];
  const float*     bl   = (const float*)d_in[3];
  const float*     Wr   = (const float*)d_in[4];
  const float*     Wout = (const float*)d_in[5];
  const float*     bout = (const float*)d_in[6];

  const int D = in_sizes[5];              // 1024
  const int N = in_sizes[0] / D;          // 50000
  const int E = in_sizes[1] / 2;          // 400000
  const int L = in_sizes[3] / D;          // 3
  const long long Npad = (((long long)N + 127) / 128) * 128;

  // bump allocator over d_ws (256B aligned sub-buffers)
  char* p = (char*)d_ws;
  auto take = [&](size_t bytes) {
    char* r = p;
    p += (bytes + 255) & ~(size_t)255;
    return r;
  };
  unsigned short* wl_bf = (unsigned short*)take((size_t)L * D * D * 2);
  unsigned short* wr_bf = (unsigned short*)take((size_t)L * D * D * 2);
  float* deg = (float*)take((size_t)N * 4);
  float* agg = (float*)take((size_t)N * D * 4);
  unsigned short* mbf = (unsigned short*)take((size_t)Npad * D * 2);
  unsigned short* hbf = (unsigned short*)take((size_t)Npad * D * 2);
  float* h0 = (float*)take((size_t)N * D * 4);
  float* h1 = (float*)take((size_t)N * D * 4);

  const long long* srcI = ei;
  const long long* dstI = ei + E;

  // weights -> bf16 (recomputed every call: deterministic, no cached state)
  {
    const long long nW = (long long)L * D * D;
    const int blocks = (int)((nW + 255) / 256);
    k_f32_to_bf16<<<blocks, 256, 0, stream>>>(Wl, wl_bf, nW);
    k_f32_to_bf16<<<blocks, 256, 0, stream>>>(Wr, wr_bf, nW);
  }

  // degrees (edge structure is layer-invariant)
  (void)hipMemsetAsync(deg, 0, (size_t)N * 4, stream);
  k_degree<<<(E + 255) / 256, 256, 0, stream>>>(dstI, deg, E);

  const float* cur = x;
  float* outs[2] = {h0, h1};
  dim3 ggrid(D / 128, (unsigned)(Npad / 128));
  for (int l = 0; l < L; ++l) {
    float* nxt = outs[l & 1];
    (void)hipMemsetAsync(agg, 0, (size_t)N * D * 4, stream);
    k_scatter_add<<<E, D / 4, 0, stream>>>(cur, srcI, dstI, agg, D);
    k_mean_convert<<<(unsigned)Npad, D / 4, 0, stream>>>(agg, deg, cur, mbf,
                                                         hbf, N, D);
    k_sage_gemm<<<ggrid, 256, 0, stream>>>(
        mbf, wl_bf + (size_t)l * D * D, hbf, wr_bf + (size_t)l * D * D,
        bl + (size_t)l * D, nxt, N, D, D);
    cur = nxt;
  }

  // final D -> 1 projection
  {
    const long long threads = (long long)N * 32;
    k_outproj<<<(unsigned)((threads + 255) / 256), 256, 0, stream>>>(
        cur, Wout, bout, (float*)d_out, N, D);
  }
}
